// TemporalGNN_43198781063869
// MI455X (gfx1250) — matrix operand order repaired and checked by reading, not compiled, not run on hardware
//
#include <hip/hip_runtime.h>
#include <hip/hip_bf16.h>

// ---------------------------------------------------------------------------
// A3TGCN (TemporalGNN) for MI455X / gfx1250, wave32 + WMMA f16 (f32 accum).
//
// Algebraic restructure: gcn(x,W,b) = (Agg x) @ W + b where Agg is the
// normalized adjacency. Agg is applied ONCE to the raw [N,8,12] features
// (96 floats/node payload, L2-resident), shared by all 3 gates & 12 periods.
// All dense GEMMs run on v_wmma_f32_16x16x32_f16 with weights pre-swizzled
// into LDS B-fragment layout.
// ---------------------------------------------------------------------------

typedef __attribute__((ext_vector_type(16))) _Float16 v16h;
typedef __attribute__((ext_vector_type(8)))  float    v8f;

#define WMMA16(a,b,c) \
  __builtin_amdgcn_wmma_f32_16x16x32_f16(false,(a),false,(b),(short)0,(c),false,false)

#define FEAT   96   // IN_CH * PERIODS
#define NPER   12
#define WAVES  8

// ------------------------------ graph prep --------------------------------

__global__ __launch_bounds__(256) void k_init_deg(float* deg, int N) {
  int i = blockIdx.x * blockDim.x + threadIdx.x;
  if (i < N) deg[i] = 1.0f;                       // self loop
}

__global__ __launch_bounds__(256) void k_count_deg(const int* ei, float* deg, int E) {
  int i = blockIdx.x * blockDim.x + threadIdx.x;
  if (i < E) atomicAdd(&deg[ei[E + i]], 1.0f);    // dst = row 1
}

__global__ __launch_bounds__(256) void k_dinv(float* deg, int N) {
  int i = blockIdx.x * blockDim.x + threadIdx.x;
  if (i < N) deg[i] = rsqrtf(deg[i]);
}

// self-loop contribution: agg[i] = x[i] * dinv[node]^2
__global__ __launch_bounds__(256) void k_agg_init(const float* __restrict__ x,
                                                  const float* __restrict__ dinv,
                                                  float* __restrict__ agg, int N) {
  int i = blockIdx.x * blockDim.x + threadIdx.x;
  if (i < N * FEAT) {
    float d = dinv[i / FEAT];
    agg[i] = x[i] * d * d;
  }
}

// one wave per edge: gather 96 floats from src (coalesced), atomic-add to dst
__global__ __launch_bounds__(256) void k_agg_edges(const float* __restrict__ x,
                                                   const int* __restrict__ ei,
                                                   const float* __restrict__ dinv,
                                                   float* __restrict__ agg, int E) {
  int gid  = blockIdx.x * blockDim.x + threadIdx.x;
  int e    = gid >> 5;
  int lane = gid & 31;
  if (e >= E) return;
  int s = ei[e];
  int d = ei[E + e];
  float nrm = dinv[s] * dinv[d];
  const float* xs = x + (long long)s * FEAT;
  float*       ad = agg + (long long)d * FEAT;
#pragma unroll
  for (int k = 0; k < 3; ++k) {
    int c = lane + 32 * k;
    atomicAdd(&ad[c], xs[c] * nrm);
  }
}

// --------------------------- fragment helpers -----------------------------

// B fragment layout (32x16 f16): lane&15 = column N, lane&16 selects K offset 8,
// vector element j follows ISA table order (j<8: k=8h+j, else 16+8h+j-8). Staged contiguously per lane.
__device__ inline void stage_frags(const float* __restrict__ W, _Float16* frag,
                                   int K, int stride, int ncols, int CT, int KC,
                                   int tid, int nt) {
  int total = CT * KC * 512;
  for (int idx = tid; idx < total; idx += nt) {
    int j    = idx & 15;
    int lane = (idx >> 4) & 31;
    int f    = idx >> 9;
    int kc   = f % KC;
    int ct   = f / KC;
    int n = ct * 16 + (lane & 15);
    int k = kc * 32 + ((lane & 16) ? 8 : 0) + ((j < 8) ? j : (16 + j - 8));
    float v = (k < K && n < ncols) ? W[k * stride + n] : 0.0f;
    frag[idx] = (_Float16)v;
  }
}

// A fragment (16x32 f16) from row-major LDS tile: lane&15 = row M,
// lane&16 selects K-half; element pairs per ISA table.
__device__ inline v16h ld_afrag(const _Float16* base, int lane, int astride, int kbase) {
  int m   = lane & 15;
  int off = (lane & 16) ? 8 : 0;
  const _Float16* row = base + m * astride + kbase + off;
  v16h a;
#pragma unroll
  for (int i = 0; i < 8; ++i) {
    int kk = (i < 4) ? (2 * i) : (16 + 2 * (i - 4));
    a[2 * i]     = row[kk];
    a[2 * i + 1] = row[kk + 1];
  }
  return a;
}

__device__ inline v16h ld_bfrag(const _Float16* frag, int f, int lane) {
  return *(const v16h*)(frag + f * 512 + lane * 16);
}

// D += A(16xK) * B(Kx64), 4 column tiles, C init = bias[n]
template <int KC>
__device__ inline void gemm_tiles(const _Float16* A, int astride,
                                  const _Float16* frag, const float* bias,
                                  int lane, v8f acc[4]) {
  int col = lane & 15;
#pragma unroll
  for (int ct = 0; ct < 4; ++ct) {
    float b = bias[ct * 16 + col];
#pragma unroll
    for (int r = 0; r < 8; ++r) acc[ct][r] = b;
  }
#pragma unroll
  for (int kc = 0; kc < KC; ++kc) {
    v16h a = ld_afrag(A, lane, astride, kc * 32);
#pragma unroll
    for (int ct = 0; ct < 4; ++ct) {
      v16h b = ld_bfrag(frag, ct * KC + kc, lane);
      acc[ct] = WMMA16(a, b, acc[ct]);
    }
  }
}

// store D-layout (16x64) into LDS tile as f16 at column base
__device__ inline void st_d16(_Float16* cb, const v8f acc[4], int lane, int colbase) {
  int col = lane & 15;
  int ro  = (lane & 16) ? 8 : 0;
#pragma unroll
  for (int ct = 0; ct < 4; ++ct)
#pragma unroll
    for (int r = 0; r < 8; ++r)
      cb[(r + ro) * 128 + colbase + ct * 16 + col] = (_Float16)acc[ct][r];
}

__device__ inline void sigmoid4(v8f a[4]) {
#pragma unroll
  for (int ct = 0; ct < 4; ++ct)
#pragma unroll
    for (int r = 0; r < 8; ++r) a[ct][r] = 1.0f / (1.0f + __expf(-a[ct][r]));
}

__device__ inline void tanh4(v8f a[4]) {
#pragma unroll
  for (int ct = 0; ct < 4; ++ct)
#pragma unroll
    for (int r = 0; r < 8; ++r) a[ct][r] = tanhf(a[ct][r]);
}

// ------------------------------ GRU + head --------------------------------

__global__ __launch_bounds__(256) void k_rnn(
    const float* __restrict__ agg,
    const float* __restrict__ Wz, const float* __restrict__ bz,
    const float* __restrict__ Wr, const float* __restrict__ br,
    const float* __restrict__ Wh, const float* __restrict__ bh,
    const float* __restrict__ lzW, const float* __restrict__ lzb,
    const float* __restrict__ lrW, const float* __restrict__ lrb,
    const float* __restrict__ lhW, const float* __restrict__ lhb,
    const float* __restrict__ att,
    const float* __restrict__ l1W, const float* __restrict__ l1b,
    const float* __restrict__ outW, const float* __restrict__ outb,
    float* __restrict__ out, int N, int NTILES) {
  // ---- LDS: pre-swizzled B fragments + staging buffers (~112 KB) ----
  __shared__ __align__(32) _Float16 fZ[16 * 512];    // lzW  128x64, 4kc x 4ct
  __shared__ __align__(32) _Float16 fR[16 * 512];    // lrW
  __shared__ __align__(32) _Float16 fH[16 * 512];    // lhW
  __shared__ __align__(32) _Float16 fXz[4 * 512];    // Wz 8x64 (K padded to 32)
  __shared__ __align__(32) _Float16 fXr[4 * 512];
  __shared__ __align__(32) _Float16 fXh[4 * 512];
  __shared__ __align__(32) _Float16 f1[8 * 512];     // l1W 64x64, 2kc x 4ct
  __shared__ __align__(32) _Float16 fO[2 * 512];     // outW 64x12 (N padded 16)
  __shared__ float sb[464];                          // all biases
  __shared__ __align__(32) _Float16 cbuf[WAVES * 16 * 128];  // [gcn | H] per wave
  __shared__ __align__(32) _Float16 cx[WAVES * 16 * 32];     // agg A-tile per wave

  int tid  = threadIdx.x;
  int lane = tid & 31;
  int wid  = tid >> 5;

  stage_frags(lzW, fZ, 128, 64, 64, 4, 4, tid, 256);
  stage_frags(lrW, fR, 128, 64, 64, 4, 4, tid, 256);
  stage_frags(lhW, fH, 128, 64, 64, 4, 4, tid, 256);
  stage_frags(Wz, fXz, 8, 64, 64, 4, 1, tid, 256);
  stage_frags(Wr, fXr, 8, 64, 64, 4, 1, tid, 256);
  stage_frags(Wh, fXh, 8, 64, 64, 4, 1, tid, 256);
  stage_frags(l1W, f1, 64, 64, 64, 4, 2, tid, 256);
  stage_frags(outW, fO, 64, 12, 12, 1, 2, tid, 256);
  for (int i = tid; i < 464; i += 256) {
    float v;
    if      (i < 64)  v = bz[i];
    else if (i < 128) v = br[i - 64];
    else if (i < 192) v = bh[i - 128];
    else if (i < 256) v = lzb[i - 192];
    else if (i < 320) v = lrb[i - 256];
    else if (i < 384) v = lhb[i - 320];
    else if (i < 448) v = l1b[i - 384];
    else { int j = i - 448; v = (j < NPER) ? outb[j] : 0.0f; }
    sb[i] = v;
  }
  __syncthreads();

  int tile = blockIdx.x * WAVES + wid;
  if (tile >= NTILES) return;

  _Float16* cb = cbuf + wid * (16 * 128);
  _Float16* cxw = cx + wid * (16 * 32);

  // zero-fill K-padding columns 8..31 of the input A-tile (once)
  {
    int m = lane & 15, half = lane >> 4;
#pragma unroll
    for (int j = 0; j < 16; ++j) {
      int c = half * 16 + j;
      if (c >= 8) cxw[m * 32 + c] = (_Float16)0.0f;
    }
  }

  // softmax over attention weights (12 values, per-thread)
  float av[NPER];
  float mx = -1e30f;
#pragma unroll
  for (int i = 0; i < NPER; ++i) { av[i] = att[i]; mx = fmaxf(mx, av[i]); }
  float ssum = 0.0f;
#pragma unroll
  for (int i = 0; i < NPER; ++i) { av[i] = __expf(av[i] - mx); ssum += av[i]; }
  float sinv = 1.0f / ssum;
#pragma unroll
  for (int i = 0; i < NPER; ++i) av[i] *= sinv;

  v8f Hc[4], Ac[4];
#pragma unroll
  for (int ct = 0; ct < 4; ++ct)
#pragma unroll
    for (int r = 0; r < 8; ++r) { Hc[ct][r] = 0.0f; Ac[ct][r] = 0.0f; }

  const float* aggt = agg + (long long)tile * 16 * FEAT;
  int col = lane & 15;
  int ro  = (lane & 16) ? 8 : 0;

  for (int p = 0; p < NPER; ++p) {
    // aggregated inputs (16x8) into the A-tile
    if (lane < 16) {
      int node = tile * 16 + lane;
#pragma unroll
      for (int c = 0; c < 8; ++c) {
        float v = (node < N) ? aggt[lane * FEAT + c * NPER + p] : 0.0f;
        cxw[lane * 32 + c] = (_Float16)v;
      }
    }
    // H -> columns 64..127 of concat tile
#pragma unroll
    for (int ct = 0; ct < 4; ++ct)
#pragma unroll
      for (int r = 0; r < 8; ++r)
        cb[(r + ro) * 128 + 64 + ct * 16 + col] = (_Float16)Hc[ct][r];

    v8f G[4], Zc[4], Rc[4], Tc[4];
    // ---- z gate ----
    gemm_tiles<1>(cxw, 32, fXz, sb + 0, lane, G);       // gcn_z = Agg@Wz + bz
    st_d16(cb, G, lane, 0);
    gemm_tiles<4>(cb, 128, fZ, sb + 192, lane, Zc);     // [gz|H]@lzW + lzb
    sigmoid4(Zc);
    // ---- r gate ----
    gemm_tiles<1>(cxw, 32, fXr, sb + 64, lane, G);
    st_d16(cb, G, lane, 0);
    gemm_tiles<4>(cb, 128, fR, sb + 256, lane, Rc);
    sigmoid4(Rc);
    // ---- candidate ----
    gemm_tiles<1>(cxw, 32, fXh, sb + 128, lane, G);
    st_d16(cb, G, lane, 0);
#pragma unroll
    for (int ct = 0; ct < 4; ++ct)                      // R*H -> cols 64..127
#pragma unroll
      for (int r = 0; r < 8; ++r)
        cb[(r + ro) * 128 + 64 + ct * 16 + col] = (_Float16)(Rc[ct][r] * Hc[ct][r]);
    gemm_tiles<4>(cb, 128, fH, sb + 320, lane, Tc);
    tanh4(Tc);
    // ---- state update + attention accumulation ----
    float pw = av[p];
#pragma unroll
    for (int ct = 0; ct < 4; ++ct)
#pragma unroll
      for (int r = 0; r < 8; ++r) {
        float z = Zc[ct][r];
        float h = z * Hc[ct][r] + (1.0f - z) * Tc[ct][r];
        Hc[ct][r] = h;
        Ac[ct][r] += pw * h;
      }
  }

  // ---- head: relu -> l1 -> relu -> out ----
#pragma unroll
  for (int ct = 0; ct < 4; ++ct)
#pragma unroll
    for (int r = 0; r < 8; ++r) Ac[ct][r] = fmaxf(Ac[ct][r], 0.0f);
  st_d16(cb, Ac, lane, 0);

  v8f H2[4];
  gemm_tiles<2>(cb, 128, f1, sb + 384, lane, H2);
#pragma unroll
  for (int ct = 0; ct < 4; ++ct)
#pragma unroll
    for (int r = 0; r < 8; ++r) H2[ct][r] = fmaxf(H2[ct][r], 0.0f);
  st_d16(cb, H2, lane, 0);

  v8f O;
  {
    float b = sb[448 + col];
#pragma unroll
    for (int r = 0; r < 8; ++r) O[r] = b;
#pragma unroll
    for (int kc = 0; kc < 2; ++kc) {
      v16h a  = ld_afrag(cb, lane, 128, kc * 32);
      v16h bf = ld_bfrag(fO, kc, lane);
      O = WMMA16(a, bf, O);
    }
  }
  if (col < NPER) {
#pragma unroll
    for (int r = 0; r < 8; ++r) {
      int node = tile * 16 + r + ro;
      if (node < N) out[node * NPER + col] = O[r];
    }
  }
}

// ------------------------------- launcher ---------------------------------

extern "C" void kernel_launch(void* const* d_in, const int* in_sizes, int n_in,
                              void* d_out, int out_size, void* d_ws, size_t ws_size,
                              hipStream_t stream) {
  (void)n_in; (void)out_size; (void)ws_size;
  const float* x    = (const float*)d_in[0];
  const int*   ei   = (const int*)d_in[1];   // edge_index [2,E] (integer)
  const float* Wz   = (const float*)d_in[2];
  const float* bz   = (const float*)d_in[3];
  const float* Wr   = (const float*)d_in[4];
  const float* br   = (const float*)d_in[5];
  const float* Wh   = (const float*)d_in[6];
  const float* bh   = (const float*)d_in[7];
  const float* lzW  = (const float*)d_in[8];
  const float* lzb  = (const float*)d_in[9];
  const float* lrW  = (const float*)d_in[10];
  const float* lrb  = (const float*)d_in[11];
  const float* lhW  = (const float*)d_in[12];
  const float* lhb  = (const float*)d_in[13];
  const float* att  = (const float*)d_in[14];
  const float* l1W  = (const float*)d_in[15];
  const float* l1b  = (const float*)d_in[16];
  const float* outW = (const float*)d_in[17];
  const float* outb = (const float*)d_in[18];
  float* out = (float*)d_out;

  int N = in_sizes[0] / FEAT;   // 50000
  int E = in_sizes[1] / 2;      // 1600000

  float* deg = (float*)d_ws;                        // N floats (dinv in place)
  float* agg = deg + ((N + 255) & ~255);            // N*96 floats

  k_init_deg<<<(N + 255) / 256, 256, 0, stream>>>(deg, N);
  k_count_deg<<<(E + 255) / 256, 256, 0, stream>>>(ei, deg, E);
  k_dinv<<<(N + 255) / 256, 256, 0, stream>>>(deg, N);
  k_agg_init<<<(N * FEAT + 255) / 256, 256, 0, stream>>>(x, deg, agg, N);
  long long tot = (long long)E * 32;
  k_agg_edges<<<(int)((tot + 255) / 256), 256, 0, stream>>>(x, ei, deg, agg, E);

  int NTILES = (N + 15) / 16;
  k_rnn<<<(NTILES + WAVES - 1) / WAVES, 256, 0, stream>>>(
      agg, Wz, bz, Wr, br, Wh, bh, lzW, lzb, lrW, lrb, lhW, lhb, att,
      l1W, l1b, outW, outb, out, N, NTILES);
}
